// RankingLoss_901943132458
// MI455X (gfx1250) — matrix operand order repaired and checked by reading, not compiled, not run on hardware
//
#include <hip/hip_runtime.h>
#include <hip/hip_bf16.h>

typedef __attribute__((ext_vector_type(16))) __bf16 v16bf;
typedef __attribute__((ext_vector_type(8)))  float  v8f;

#define N_ROWS     8192
#define T_COLS     128
#define INV_SIGMA  10.0f

#define WG_THREADS    128   // 4 waves (wave32)
#define ROWS_PER_WAVE 16
#define ROWS_PER_WG   64
#define LDS_STRIDE    130   // bf16 elems per row; 130*2B -> bank = row index, conflict-free gathers

__global__ void rk_zeroE_kernel(float* __restrict__ E) {
    const int t = threadIdx.x;
    if (t < T_COLS) E[t] = 0.0f;
}

__global__ __launch_bounds__(WG_THREADS)
void rk_phaseA_kernel(const float* __restrict__ logits,
                      const int*   __restrict__ dur,
                      const float* __restrict__ events,
                      float* __restrict__ E,     // [128] global accumulator
                      float* __restrict__ U)     // [N]   per-row u_i
{
    __shared__ __bf16 hiL[ROWS_PER_WG * LDS_STRIDE];
    __shared__ __bf16 loL[ROWS_PER_WG * LDS_STRIDE];
    __shared__ float  Eloc[T_COLS];
    __shared__ int    dL[ROWS_PER_WG];
    __shared__ float  evL[ROWS_PER_WG];

    const int tid  = threadIdx.x;
    const int wave = tid >> 5;
    const int lane = tid & 31;
    const int wgRowBase   = blockIdx.x * ROWS_PER_WG;
    const int waveRowBase = wgRowBase + wave * ROWS_PER_WAVE;

    if (tid < T_COLS) Eloc[tid] = 0.0f;
    if (tid < ROWS_PER_WG) {
        dL[tid]  = dur[wgRowBase + tid];
        evL[tid] = events[wgRowBase + tid];
    }

    // ---- Stage 1: softmax per row; write pmf as bf16 hi+lo into LDS ----
    for (int r = 0; r < ROWS_PER_WAVE; ++r) {
        const int row = waveRowBase + r;
        const float4 v = reinterpret_cast<const float4*>(logits + (size_t)row * T_COLS)[lane];
        float m = fmaxf(fmaxf(v.x, v.y), fmaxf(v.z, v.w));
        #pragma unroll
        for (int off = 16; off > 0; off >>= 1) m = fmaxf(m, __shfl_xor(m, off, 32));
        const float e0 = __expf(v.x - m), e1 = __expf(v.y - m);
        const float e2 = __expf(v.z - m), e3 = __expf(v.w - m);
        float s = (e0 + e1) + (e2 + e3);
        #pragma unroll
        for (int off = 16; off > 0; off >>= 1) s += __shfl_xor(s, off, 32);
        const float inv = 1.0f / s;
        const int lr = wave * ROWS_PER_WAVE + r;
        __bf16* hp = &hiL[lr * LDS_STRIDE + lane * 4];
        __bf16* lp = &loL[lr * LDS_STRIDE + lane * 4];
        const float p[4] = {e0 * inv, e1 * inv, e2 * inv, e3 * inv};
        #pragma unroll
        for (int k = 0; k < 4; ++k) {
            const __bf16 h = (__bf16)p[k];
            hp[k] = h;
            lp[k] = (__bf16)(p[k] - (float)h);
        }
    }
    __syncthreads();

    // ---- Stage 2: gather A tiles in ISA 16-bit A-matrix 16x32 layout ----
    // lane L: M = L&15; half = L>>4; vgpr v holds K pairs:
    //   v<4 : K = 2v+pos + 8*half ;  v>=4 : K = 16 + 2(v-4)+pos + 8*half
    const int m    = lane & 15;
    const int half = lane >> 4;
    const int lrA  = wave * ROWS_PER_WAVE + m;
    v16bf ahi[4], alo[4];
    #pragma unroll
    for (int kt = 0; kt < 4; ++kt) {
        #pragma unroll
        for (int e = 0; e < 16; ++e) {
            const int vv = e >> 1, pos = e & 1;
            const int K  = kt * 32 + (vv < 4 ? 2 * vv + pos : 16 + 2 * (vv - 4) + pos) + 8 * half;
            ahi[kt][e] = hiL[lrA * LDS_STRIDE + K];
            alo[kt][e] = loL[lrA * LDS_STRIDE + K];
        }
    }

    // Hoist per-row metadata for this lane's 8 output rows into registers.
    int   dReg[8];
    float evReg[8];
    #pragma unroll
    for (int vv = 0; vv < 8; ++vv) {
        const int lr = wave * ROWS_PER_WAVE + vv + 8 * half;
        dReg[vv]  = dL[lr];
        evReg[vv] = evL[lr];
    }

    // ---- Stage 3: cdf = pmf @ upper-triangular-ones via WMMA, fused epilogue ----
    // B layout (32x16 bf16): lane L -> N = L&15, element e -> K = (e<8 ? e : e+8) + 8*half
    // C/D layout: element v -> row M = v + 8*half, col N = L&15
    const int n = lane & 15;
    float aAcc[8] = {0.f, 0.f, 0.f, 0.f, 0.f, 0.f, 0.f, 0.f};

    for (int ct = 0; ct < 8; ++ct) {
        v8f acc = {};
        const int t = ct * 16 + n;
        #pragma unroll
        for (int kt = 0; kt < 4; ++kt) {
            if (kt * 32 > ct * 16 + 15) break;            // all-zero B tile (triangular skip)
            const int lim = t - kt * 32 - 8 * half;       // b[e] = (K <= t) <=> ((e<8?e:e+8) <= lim)
            v16bf b;
            #pragma unroll
            for (int e = 0; e < 16; ++e)
                b[e] = ((e < 8 ? e : e + 8) <= lim) ? (__bf16)1.0f : (__bf16)0.0f;
            acc = __builtin_amdgcn_wmma_f32_16x16x32_bf16(false, alo[kt], false, b,
                                                          (short)0, acc, false, false);
            acc = __builtin_amdgcn_wmma_f32_16x16x32_bf16(false, ahi[kt], false, b,
                                                          (short)0, acc, false, false);
        }
        float part = 0.0f;
        #pragma unroll
        for (int vv = 0; vv < 8; ++vv) {
            const float c  = acc[vv];                     // cdf[row, t]
            const int   dr = dReg[vv];
            const float ex = __expf(c * INV_SIGMA);       // branchless: select below
            part     += (dr > t)  ? ex : 0.0f;
            aAcc[vv] += (dr == t) ? c  : 0.0f;            // exactly one (n,ct) hit per row
        }
        atomicAdd(&Eloc[t], part);
    }

    // Reduce a_i across the 16 lanes sharing each row (xor of lane bits 0..3
    // stays within a half), then two lanes (n==0) store all 8 rows.
    float aRed[8];
    #pragma unroll
    for (int vv = 0; vv < 8; ++vv) {
        float a = aAcc[vv];
        a += __shfl_xor(a, 1, 32);
        a += __shfl_xor(a, 2, 32);
        a += __shfl_xor(a, 4, 32);
        a += __shfl_xor(a, 8, 32);
        aRed[vv] = a;
    }
    if (n == 0) {
        #pragma unroll
        for (int vv = 0; vv < 8; ++vv) {
            const int lr = wave * ROWS_PER_WAVE + vv + 8 * half;
            U[wgRowBase + lr] = evReg[vv] * __expf(-aRed[vv] * INV_SIGMA);
        }
    }
    __syncthreads();
    if (tid < T_COLS) atomicAdd(&E[tid], Eloc[tid]);
}

__global__ __launch_bounds__(256)
void rk_phaseB_kernel(const float* __restrict__ E,
                      const float* __restrict__ U,
                      const int*   __restrict__ dur,
                      float* __restrict__ out)
{
    __shared__ float Esh[T_COLS];
    __shared__ float red[8];
    const int tid = threadIdx.x;
    if (tid < T_COLS) Esh[tid] = E[tid];
    __syncthreads();
    float s = 0.0f;
    for (int i = tid; i < N_ROWS; i += 256)
        s += U[i] * Esh[dur[i] & (T_COLS - 1)];
    #pragma unroll
    for (int off = 16; off > 0; off >>= 1) s += __shfl_xor(s, off, 32);
    if ((tid & 31) == 0) red[tid >> 5] = s;
    __syncthreads();
    if (tid == 0) {
        float tot = 0.0f;
        #pragma unroll
        for (int w = 0; w < 8; ++w) tot += red[w];
        out[0] = tot / ((float)N_ROWS * (float)N_ROWS);
    }
}

extern "C" void kernel_launch(void* const* d_in, const int* in_sizes, int n_in,
                              void* d_out, int out_size, void* d_ws, size_t ws_size,
                              hipStream_t stream) {
    const float* logits = (const float*)d_in[0];
    const int*   dur    = (const int*)d_in[1];
    const float* events = (const float*)d_in[2];
    float* E = (float*)d_ws;          // 128 floats
    float* U = E + T_COLS;            // N floats

    rk_zeroE_kernel<<<1, 128, 0, stream>>>(E);
    rk_phaseA_kernel<<<N_ROWS / ROWS_PER_WG, WG_THREADS, 0, stream>>>(logits, dur, events, E, U);
    rk_phaseB_kernel<<<1, 256, 0, stream>>>(E, U, dur, (float*)d_out);
}
